// MEGATHeadConv_90331752169537
// MI455X (gfx1250) — compile-verified, matrix operand order checked by reading, not run on hardware
//
#include <hip/hip_runtime.h>
#include <hip/hip_bf16.h>
#include <math.h>

typedef __bf16 bf16_t;
typedef __attribute__((ext_vector_type(16))) __bf16 v16bf;
typedef __attribute__((ext_vector_type(8)))  float  v8f;

#define NN      128          // nodes
#define DM      256          // feature dim
#define BATCH   8
#define NEDGE   (NN*NN)                  // 16384 edges per batch
#define NROWS_NODE (BATCH*NN)            // 1024
#define NROWS_EDGE (BATCH*NEDGE)         // 131072
#define BN_EPS  1e-5f
#define LDS_PITCH 264                    // halves per staged row (256 + 8 pad)

// K index of half-pair base for VGPR v (0..7) within a 16x32 bf16 A/B fragment.
// A layout (ISA 7.12.2): lanes 0-15 hold K {0..7,16..23}; lanes 16-31 K {8..15,24..31},
// two K per VGPR. B assumed lane-dual (lane = N column).
__device__ __forceinline__ int kOfV(int v, int lane) {
    return 2 * (v & 3) + ((v >= 4) ? 16 : 0) + (((lane & 31) >= 16) ? 8 : 0);
}

// ---------------------------------------------------------------------------
// Pack a 256x256 weight (row-major, leading dim ldw; value used = W[n*ldw+k])
// into per-lane B fragments: out[((ks*16+tile)*32+lane)*16 + i] = bf16(W[n][k])
// ---------------------------------------------------------------------------
__global__ void pack_w256(const float* __restrict__ W, int ldw, bf16_t* __restrict__ out) {
    int t    = blockIdx.x * 256 + threadIdx.x;   // 65536 elements
    int i    = t & 15;
    int lane = (t >> 4) & 31;
    int tile = (t >> 9) & 15;
    int ks   = t >> 13;
    int v    = i >> 1;
    int k    = ks * 32 + kOfV(v, lane) + (i & 1);
    int n    = tile * 16 + (lane & 15);
    out[t]   = (bf16_t)W[(long)n * ldw + k];
}

// M[o][d] = sum_c W_proj[o][512+c] * W_e[c][d]   (256x256, plain f32 VALU)
__global__ void compute_M(const float* __restrict__ Wproj, const float* __restrict__ We,
                          float* __restrict__ M) {
    int o = blockIdx.x, d = threadIdx.x;
    const float* wz = Wproj + (long)o * 768 + 512;
    float acc = 0.f;
    for (int c = 0; c < 256; ++c) acc += wz[c] * We[(long)c * 256 + d];
    M[(long)o * 256 + d] = acc;
}

// v[d] = sum_c W_attn[512+c] * W_e[c][d]
__global__ void compute_v(const float* __restrict__ Wattn, const float* __restrict__ We,
                          float* __restrict__ v) {
    int d = threadIdx.x;
    float acc = 0.f;
    for (int c = 0; c < 256; ++c) acc += Wattn[512 + c] * We[(long)c * 256 + d];
    v[d] = acc;
}

// a_s[row] = z_h[row] . W_attn[0:256] ; a_e[row] = z_h[row] . W_attn[256:512]
__global__ void attn_se(const float* __restrict__ zh, const float* __restrict__ Wattn,
                        float* __restrict__ a_s, float* __restrict__ a_e) {
    int row = blockIdx.x * 128 + threadIdx.x;    // 1024 rows
    const float* z = zh + (long)row * DM;
    float s = 0.f, e = 0.f;
    for (int c = 0; c < 256; ++c) { float zv = z[c]; s += zv * Wattn[c]; e += zv * Wattn[256 + c]; }
    a_s[row] = s; a_e[row] = e;
}

// ---------------------------------------------------------------------------
// Core: 16 rows x (4x16) cols, K=256, bf16 WMMA with f32 accumulate.
// Each wave computes N-tiles [tbase, tbase+4). As = staged LDS tile
// [16][LDS_PITCH] bf16 shared by all waves of the block; Bp = packed fragments.
// Per-wave regs: 4x v8f acc + 8 A + 8 B -> no spills.
// ---------------------------------------------------------------------------
__device__ __forceinline__ void wmma_core4(const bf16_t* As, const bf16_t* __restrict__ Bp,
                                           int tbase, v8f acc[4]) {
    int lane = threadIdx.x & 31;
    int m    = lane & 15;
    #pragma unroll
    for (int ks = 0; ks < 8; ++ks) {
        union { v16bf v; unsigned u[8]; } a;
        #pragma unroll
        for (int vv = 0; vv < 8; ++vv) {
            int k = ks * 32 + kOfV(vv, lane);               // even -> dword aligned
            a.u[vv] = *(const unsigned*)(As + m * LDS_PITCH + k);
        }
        #pragma unroll
        for (int tt = 0; tt < 4; ++tt) {
            union { v16bf v; uint4 q[2]; } b;
            const uint4* bp = (const uint4*)(Bp + (((long)ks * 16 + tbase + tt) * 32 + lane) * 16);
            b.q[0] = bp[0]; b.q[1] = bp[1];
            acc[tt] = __builtin_amdgcn_wmma_f32_16x16x32_bf16(
                          false, a.v, false, b.v, (short)0, acc[tt], false, false);
        }
    }
}

// Generic GEMM: C[r][n] = sum_k A[r][k] * W[n][k]; 16 rows per 4-wave block.
__global__ void gemm16_bf16(const float* __restrict__ A, const bf16_t* __restrict__ Bp,
                            float* __restrict__ C) {
    __shared__ bf16_t As[16 * LDS_PITCH];
    int  tid  = threadIdx.x;             // 128
    int  lane = tid & 31;
    int  wv   = tid >> 5;                // wave id 0..3
    long r0   = (long)blockIdx.x * 16;
    #pragma unroll
    for (int rr = 0; rr < 4; ++rr) {     // wave wv stages rows 4*wv..4*wv+3
        int r = wv * 4 + rr;
        const float4* src = (const float4*)(A + (r0 + r) * DM) + lane * 2;
        float4 f0 = src[0], f1 = src[1];
        bf16_t* dst = As + r * LDS_PITCH + lane * 8;
        dst[0] = (bf16_t)f0.x; dst[1] = (bf16_t)f0.y; dst[2] = (bf16_t)f0.z; dst[3] = (bf16_t)f0.w;
        dst[4] = (bf16_t)f1.x; dst[5] = (bf16_t)f1.y; dst[6] = (bf16_t)f1.z; dst[7] = (bf16_t)f1.w;
    }
    __syncthreads();
    v8f acc[4];
    #pragma unroll
    for (int tt = 0; tt < 4; ++tt) acc[tt] = (v8f){0,0,0,0,0,0,0,0};
    wmma_core4(As, Bp, wv * 4, acc);
    int mbase = (lane < 16) ? 0 : 8;
    #pragma unroll
    for (int tt = 0; tt < 4; ++tt) {
        int col = (wv * 4 + tt) * 16 + (lane & 15);
        #pragma unroll
        for (int vv = 0; vv < 8; ++vv)
            C[(r0 + mbase + vv) * DM + col] = acc[tt][vv];
    }
}

// ---------------------------------------------------------------------------
// Edge megakernel: z_e_proj[b,e,:] = edge@M^T + Ps[j] + Pe[i]; attn logit fused.
// 16 edges per 4-wave block. 8192 blocks.
// ---------------------------------------------------------------------------
__global__ void edge_main(const float* __restrict__ edge, const bf16_t* __restrict__ Mpack,
                          const float* __restrict__ vvec,
                          const float* __restrict__ Ps, const float* __restrict__ Pe,
                          const float* __restrict__ a_s, const float* __restrict__ a_e,
                          float* __restrict__ zeproj, float* __restrict__ attnlog) {
    __shared__ bf16_t As[16 * LDS_PITCH];
    __shared__ float  lred[16];
    int  tid  = threadIdx.x;             // 128
    int  lane = tid & 31;
    int  wv   = tid >> 5;
    long R0   = (long)blockIdx.x * 16;   // global edge row
    int  bix  = (int)(R0 >> 14);         // /16384
    int  e0   = (int)(R0 & (NEDGE - 1));
    if (tid < 16) lred[tid] = 0.f;
    __syncthreads();
    float4 v0 = ((const float4*)vvec)[lane * 2];
    float4 v1 = ((const float4*)vvec)[lane * 2 + 1];
    const float* eb = edge + (long)bix * NEDGE * DM;
    #pragma unroll
    for (int rr = 0; rr < 4; ++rr) {
        int r = wv * 4 + rr;
        const float4* src = (const float4*)(eb + (long)(e0 + r) * DM) + lane * 2;
        float4 f0 = src[0], f1 = src[1];
        float  d  = f0.x*v0.x + f0.y*v0.y + f0.z*v0.z + f0.w*v0.w
                  + f1.x*v1.x + f1.y*v1.y + f1.z*v1.z + f1.w*v1.w;
        atomicAdd(&lred[r], d);                       // ds_add_f32 (f32 attn dot)
        bf16_t* dst = As + r * LDS_PITCH + lane * 8;
        dst[0] = (bf16_t)f0.x; dst[1] = (bf16_t)f0.y; dst[2] = (bf16_t)f0.z; dst[3] = (bf16_t)f0.w;
        dst[4] = (bf16_t)f1.x; dst[5] = (bf16_t)f1.y; dst[6] = (bf16_t)f1.z; dst[7] = (bf16_t)f1.w;
    }
    __syncthreads();
    if (R0 + 16 < (long)NROWS_EDGE)
        __builtin_prefetch(edge + (R0 + 16) * DM, 0, 1);   // global_prefetch_b8
    v8f acc[4];
    #pragma unroll
    for (int tt = 0; tt < 4; ++tt) acc[tt] = (v8f){0,0,0,0,0,0,0,0};
    wmma_core4(As, Mpack, wv * 4, acc);
    int mbase = (lane < 16) ? 0 : 8;
    #pragma unroll
    for (int tt = 0; tt < 4; ++tt) {
        int col = (wv * 4 + tt) * 16 + (lane & 15);
        #pragma unroll
        for (int vv = 0; vv < 8; ++vv) {
            int e = e0 + mbase + vv;
            int i = e >> 7, j = e & 127;
            float val = acc[tt][vv]
                      + Ps[((long)bix * NN + j) * DM + col]
                      + Pe[((long)bix * NN + i) * DM + col];
            zeproj[((long)bix * NEDGE + e) * DM + col] = val;
        }
    }
    if (tid < 16) {
        int e = e0 + tid;
        int i = e >> 7, j = e & 127;
        float l = lred[tid] + a_s[bix * NN + j] + a_e[bix * NN + i];
        l = (l > 0.f) ? l : 0.01f * l;               // leaky_relu
        attnlog[(long)bix * NEDGE + e] = l;
    }
}

// masked softmax over j + z_agg[b,i,:] = sum_j attn * z_h[b,j,:]; 1 block per (b,i)
__global__ void softmax_agg(const float* __restrict__ attnlog, const float* __restrict__ adj,
                            const float* __restrict__ zh, float* __restrict__ zagg) {
    __shared__ float sh[128];
    __shared__ float wsh[128];
    int tid = threadIdx.x;                 // 256
    int row = blockIdx.x;                  // b*128 + i
    int b = row >> 7, i = row & 127;
    if (tid < 128) {
        float l = attnlog[(long)b * NEDGE + i * 128 + tid];
        float a = adj[(long)row * 128 + tid];
        l = (a < 0.5f) ? -INFINITY : l;
        wsh[tid] = l; sh[tid] = l;
    }
    __syncthreads();
    for (int s = 64; s > 0; s >>= 1) { if (tid < s) sh[tid] = fmaxf(sh[tid], sh[tid + s]); __syncthreads(); }
    float mx = fmaxf(sh[0], -1e30f);
    __syncthreads();
    if (tid < 128) { float e = __expf(wsh[tid] - mx); wsh[tid] = e; sh[tid] = e; }
    __syncthreads();
    for (int s = 64; s > 0; s >>= 1) { if (tid < s) sh[tid] += sh[tid + s]; __syncthreads(); }
    float total = sh[0];
    __syncthreads();
    if (tid < 128) wsh[tid] = wsh[tid] / total;
    __syncthreads();
    int d = tid;
    float acc = 0.f;
    #pragma unroll 4
    for (int j = 0; j < 128; ++j) acc += wsh[j] * zh[((long)b * NN + j) * DM + d];
    zagg[(long)row * DM + d] = acc;
}

// Reference _bn_train: stats over axes (0,2) of [b, slots, D] -> per-slot mean/var
__global__ void slot_stats(const float* __restrict__ y, int nslots,
                           float* __restrict__ omean, float* __restrict__ orsig) {
    __shared__ float ss[256], sq[256];
    int slot = blockIdx.x, d = threadIdx.x;
    float s = 0.f, q = 0.f;
    for (int b = 0; b < BATCH; ++b) {
        float v = y[((long)b * nslots + slot) * DM + d];
        s += v; q += v * v;
    }
    ss[d] = s; sq[d] = q; __syncthreads();
    for (int st = 128; st > 0; st >>= 1) {
        if (d < st) { ss[d] += ss[d + st]; sq[d] += sq[d + st]; }
        __syncthreads();
    }
    if (d == 0) {
        float cnt = (float)(BATCH * DM);
        float m = ss[0] / cnt;
        float var = sq[0] / cnt - m * m;
        omean[slot] = m; orsig[slot] = rsqrtf(var + BN_EPS);
    }
}

__global__ void final_x(const float* __restrict__ x, const float* __restrict__ zagg,
                        const float* __restrict__ nmean, const float* __restrict__ nrsig,
                        float* __restrict__ out) {
    long idx = (long)blockIdx.x * 256 + threadIdx.x;   // 262144
    int  i   = (int)((idx >> 8) & 127);                // node index (BN channel)
    float y  = x[idx] + (zagg[idx] - nmean[i]) * nrsig[i];
    out[idx] = (y > 0.f) ? y : (__expf(y) - 1.f);
}

__global__ void final_edge(const float* __restrict__ zeproj, const float* __restrict__ emean,
                           const float* __restrict__ ersig, float* __restrict__ out) {
    long i4 = (long)blockIdx.x * 256 + threadIdx.x;    // float4 index
    int  e  = (int)((i4 >> 6) & (NEDGE - 1));          // edge slot (BN channel)
    float m = emean[e], rs = ersig[e];
    float4 z = ((const float4*)zeproj)[i4];
    float zz[4] = { z.x, z.y, z.z, z.w };
    float rr[4];
    #pragma unroll
    for (int k = 0; k < 4; ++k) {
        float y = (zz[k] - m) * rs;
        rr[k] = (y > 0.f) ? y : (__expf(y) - 1.f);
    }
    float4 o; o.x = rr[0]; o.y = rr[1]; o.z = rr[2]; o.w = rr[3];
    ((float4*)out)[i4] = o;
}

// ---------------------------------------------------------------------------
extern "C" void kernel_launch(void* const* d_in, const int* in_sizes, int n_in,
                              void* d_out, int out_size, void* d_ws, size_t ws_size,
                              hipStream_t stream) {
    (void)in_sizes; (void)n_in; (void)out_size; (void)ws_size;
    const float* adj    = (const float*)d_in[0];
    const float* x      = (const float*)d_in[1];
    const float* edge   = (const float*)d_in[2];
    const float* W_h    = (const float*)d_in[3];
    const float* W_e    = (const float*)d_in[4];
    const float* W_proj = (const float*)d_in[5];
    const float* W_attn = (const float*)d_in[6];
    float* out = (float*)d_out;

    // workspace layout (float offsets)
    float* ws = (float*)d_ws;
    float* zh     = ws + 0;            // 262144
    float* Ps     = ws + 262144;       // 262144
    float* Pe     = ws + 524288;       // 262144
    float* as_    = ws + 786432;       // 1024
    float* ae_    = ws + 787456;       // 1024
    float* vv     = ws + 788480;       // 256
    float* Mf     = ws + 788736;       // 65536
    float* alog   = ws + 854272;       // 131072
    float* zagg   = ws + 985344;       // 262144
    float* emean  = ws + 1247488;      // 16384
    float* ersig  = ws + 1263872;      // 16384
    float* nmean  = ws + 1280256;      // 128
    float* nrsig  = ws + 1280384;      // 128
    float* zeproj = ws + 1280512;      // 33554432
    bf16_t* packs  = (bf16_t*)((char*)d_ws + (size_t)(1280512 + 33554432) * 4);
    bf16_t* Whpack = packs;            // 65536 bf16 each
    bf16_t* Pspack = packs + 65536;
    bf16_t* Pepack = packs + 131072;
    bf16_t* Mpack  = packs + 196608;

    // weight prep
    pack_w256<<<256, 256, 0, stream>>>(W_h, 256, Whpack);
    pack_w256<<<256, 256, 0, stream>>>(W_proj, 768, Pspack);
    pack_w256<<<256, 256, 0, stream>>>(W_proj + 256, 768, Pepack);
    compute_M<<<256, 256, 0, stream>>>(W_proj, W_e, Mf);
    compute_v<<<1, 256, 0, stream>>>(W_attn, W_e, vv);
    pack_w256<<<256, 256, 0, stream>>>(Mf, 256, Mpack);

    // node-side GEMMs (1024 rows each)
    gemm16_bf16<<<NROWS_NODE / 16, 128, 0, stream>>>(x,  Whpack, zh);
    gemm16_bf16<<<NROWS_NODE / 16, 128, 0, stream>>>(zh, Pspack, Ps);
    gemm16_bf16<<<NROWS_NODE / 16, 128, 0, stream>>>(zh, Pepack, Pe);
    attn_se<<<8, 128, 0, stream>>>(zh, W_attn, as_, ae_);

    // big fused edge pass
    edge_main<<<NROWS_EDGE / 16, 128, 0, stream>>>(edge, Mpack, vv, Ps, Pe, as_, ae_, zeproj, alog);

    // BN stats (per-slot, per the reference's axis-(0,2) reduction)
    slot_stats<<<NEDGE, 256, 0, stream>>>(zeproj, NEDGE, emean, ersig);

    // attention softmax + aggregation, node BN stats
    softmax_agg<<<NROWS_NODE, 256, 0, stream>>>(alog, adj, zh, zagg);
    slot_stats<<<NN, 256, 0, stream>>>(zagg, NN, nmean, nrsig);

    // outputs: x_out then edge_out, concatenated
    final_x<<<NROWS_NODE * DM / 256, 256, 0, stream>>>(x, zagg, nmean, nrsig, out);
    final_edge<<<(long)NROWS_EDGE * DM / 4 / 256, 256, 0, stream>>>(zeproj, emean, ersig,
                                                                   out + (long)NROWS_NODE * DM);
}
